// PanguAttention_53472342835460
// MI455X (gfx1250) — compile-verified, matrix operand order checked by reading, not run on hardware
//
#include <hip/hip_runtime.h>
#include <hip/hip_bf16.h>
#include <stdint.h>

typedef __bf16 bf16_t;
typedef __attribute__((ext_vector_type(16))) __bf16 bf16x16;
typedef __attribute__((ext_vector_type(8)))  __bf16 bf16x8;
typedef __attribute__((ext_vector_type(8)))  float  f32x8;

typedef unsigned int        u32;
typedef unsigned long long  u64;
typedef u32 u32x4v __attribute__((ext_vector_type(4)));
typedef int i32x4v __attribute__((ext_vector_type(4)));
typedef int i32x8v __attribute__((ext_vector_type(8)));

#define B_    2
#define S_    2048
#define D_    2048
#define H_    16
#define HD_   128
#define NQKV  (3 * D_)
#define NTOK  (B_ * S_)

// ---------------------------------------------------------------- helpers

__device__ __forceinline__ f32x8 zero8() {
  f32x8 z;
#pragma unroll
  for (int i = 0; i < 8; ++i) z[i] = 0.0f;
  return z;
}

// 16x32 bf16 WMMA fragment load (A layout) from row-major storage, also used
// for B fragments from transposed (column-contiguous) storage.
__device__ __forceinline__ bf16x16 frag_ld(const bf16_t* p0, int ld) {
  const int lane = threadIdx.x & 31;
  const int r    = lane & 15;
  const int kg   = (lane >> 4) << 3;
  const bf16_t* p = p0 + r * ld + kg;
  bf16x8 lo = *reinterpret_cast<const bf16x8*>(p);
  bf16x8 hi = *reinterpret_cast<const bf16x8*>(p + 16);
  bf16x16 v;
#pragma unroll
  for (int i = 0; i < 8; ++i) { v[i] = lo[i]; v[8 + i] = hi[i]; }
  return v;
}

__device__ __forceinline__ f32x8 wmma_bf16(bf16x16 a, bf16x16 b, f32x8 c) {
  return __builtin_amdgcn_wmma_f32_16x16x32_bf16(
      false, a, false, b, (short)0, c, false, false);
}

// Tensor Data Mover: async 2D tile load (global -> LDS), bf16 elements.
// pad_int/pad_amt are the encoded D# values (pad_interval: 3 => 16 DWORDs,
// 5 => 64 DWORDs; pad_amount: 3 => 4 DWORDs).
__device__ __forceinline__ void tdm_load_2d(const void* gptr, u32 lds_off,
                                            u32 dim0, u32 dim1, u32 stride0,
                                            u32 tile0, u32 tile1,
                                            u32 pad_int, u32 pad_amt) {
  u64 ga = (u64)(uintptr_t)gptr;
  u32x4v g0;
  g0[0] = 1u;                                               // count=1
  g0[1] = lds_off;                                          // LDS byte addr
  g0[2] = (u32)ga;                                          // global addr lo
  g0[3] = ((u32)(ga >> 32) & 0x01FFFFFFu) | 0x80000000u;    // addr hi, type=2

  // group1: wg_mask=0 | data_size=1(2B) | pad_enable | pad_interval | pad_amount
  u64 q0 = (u64)((1u << 16) | (1u << 20) | (pad_int << 22) | (pad_amt << 25)) |
           ((u64)(dim0 & 0xFFFFu) << 48);                   // tensor_dim0 lo
  u64 q1 = (u64)(dim0 >> 16) |                              // tensor_dim0 hi
           ((u64)dim1 << 16) |                              // tensor_dim1
           ((u64)(tile0 & 0xFFFFu) << 48);                  // tile_dim0
  u64 q2 = (u64)(tile1 & 0xFFFFu) |                         // tile_dim1
           ((u64)stride0 << 32);                            // dim0_stride lo32
  u64 q3 = 0;

  union { u64 q[4]; i32x8v v; } g1;
  g1.q[0] = q0; g1.q[1] = q1; g1.q[2] = q2; g1.q[3] = q3;

  i32x4v z4 = {0, 0, 0, 0};
  i32x8v z8 = {0, 0, 0, 0, 0, 0, 0, 0};
  __builtin_amdgcn_tensor_load_to_lds(g0, g1.v, z4, z4, z8, 0);
}

// ---------------------------------------------------------------- convert

__global__ void k_f32_to_bf16(const float* __restrict__ src,
                              bf16_t* __restrict__ dst, int n) {
  int i = blockIdx.x * blockDim.x + threadIdx.x;
  if (i < n) dst[i] = (bf16_t)src[i];
}

// src [rows][cols] fp32 -> dst [cols][rows] bf16 (transpose)
__global__ void k_f32_to_bf16_T(const float* __restrict__ src,
                                bf16_t* __restrict__ dst, int rows, int cols) {
  int c = blockIdx.x * blockDim.x + threadIdx.x;
  int r = blockIdx.y;
  if (c < cols) dst[(size_t)c * rows + r] = (bf16_t)src[(size_t)r * cols + c];
}

// ---------------------------------------------------------------- QKV GEMM
// C[NTOK, NQKV] = X[NTOK, D] * Wqkv[D, NQKV]; B supplied pre-transposed
// WT[NQKV, D]. Tiles staged by the Tensor Data Mover, double-buffered.

#define BK_    32
#define LDA_   (BK_ + 8)

__global__ __launch_bounds__(256) void k_gemm_qkv(
    const bf16_t* __restrict__ Abf, const bf16_t* __restrict__ WT,
    bf16_t* __restrict__ Qb, bf16_t* __restrict__ Kb, bf16_t* __restrict__ Vb) {
  __shared__ bf16_t As[2][128][LDA_];
  __shared__ bf16_t Bs[2][128][LDA_];

  const int t    = threadIdx.x;
  const int wid  = t >> 5;
  const int lane = t & 31;
  const int wm   = wid & 3;
  const int wn   = wid >> 2;
  const int m0   = blockIdx.y * 128;
  const int n0   = blockIdx.x * 128;
  const int wuid = __builtin_amdgcn_readfirstlane(t) >> 5;  // scalar wave id

  f32x8 acc[2][4];
#pragma unroll
  for (int mt = 0; mt < 2; ++mt)
#pragma unroll
    for (int nt = 0; nt < 4; ++nt) acc[mt][nt] = zero8();

  u32 ldsA[2] = { (u32)(uintptr_t)&As[0][0][0], (u32)(uintptr_t)&As[1][0][0] };
  u32 ldsB[2] = { (u32)(uintptr_t)&Bs[0][0][0], (u32)(uintptr_t)&Bs[1][0][0] };

  const int NIT = D_ / BK_;
  if (wuid == 0) {
    tdm_load_2d(Abf + (size_t)m0 * D_, ldsA[0], D_, NTOK - m0, D_, BK_, 128, 3, 3);
    tdm_load_2d(WT  + (size_t)n0 * D_, ldsB[0], D_, NQKV - n0, D_, BK_, 128, 3, 3);
    __builtin_amdgcn_s_wait_tensorcnt(0);
  }
  __syncthreads();

  for (int it = 0; it < NIT; ++it) {
    const int cur = it & 1;
    if (wuid == 0 && it + 1 < NIT) {
      const int k1 = (it + 1) * BK_;
      tdm_load_2d(Abf + (size_t)m0 * D_ + k1, ldsA[cur ^ 1], D_, NTOK - m0, D_,
                  BK_, 128, 3, 3);
      tdm_load_2d(WT + (size_t)n0 * D_ + k1, ldsB[cur ^ 1], D_, NQKV - n0, D_,
                  BK_, 128, 3, 3);
    }

    bf16x16 af[2], bfm[4];
#pragma unroll
    for (int mt = 0; mt < 2; ++mt)
      af[mt] = frag_ld(&As[cur][wm * 32 + mt * 16][0], LDA_);
#pragma unroll
    for (int nt = 0; nt < 4; ++nt)
      bfm[nt] = frag_ld(&Bs[cur][wn * 64 + nt * 16][0], LDA_);
#pragma unroll
    for (int mt = 0; mt < 2; ++mt)
#pragma unroll
      for (int nt = 0; nt < 4; ++nt)
        acc[mt][nt] = wmma_bf16(af[mt], bfm[nt], acc[mt][nt]);

    if (wuid == 0) __builtin_amdgcn_s_wait_tensorcnt(0);
    __syncthreads();
  }

  const u32 ncol = lane & 15;
  const u32 half = lane >> 4;
#pragma unroll
  for (int mt = 0; mt < 2; ++mt)
#pragma unroll
    for (int nt = 0; nt < 4; ++nt)
#pragma unroll
      for (int r = 0; r < 8; ++r) {
        u32 gm = (u32)m0 + wm * 32 + mt * 16 + r + 8 * half;
        u32 gn = (u32)n0 + wn * 64 + nt * 16 + ncol;
        float v = acc[mt][nt][r];
        u32 part = gn >> 11;        // / D_
        u32 c    = gn & (D_ - 1);
        u32 h    = c >> 7;          // / HD_
        u32 hd   = c & (HD_ - 1);
        u32 b    = gm >> 11;        // / S_
        u32 s    = gm & (S_ - 1);
        bf16_t* dst = (part == 0) ? Qb : ((part == 1) ? Kb : Vb);
        dst[(((size_t)b * H_ + h) * S_ + s) * HD_ + hd] = (bf16_t)v;
      }
}

// ---------------------------------------------------------------- RoPE

__global__ void k_rope(bf16_t* __restrict__ Qb, bf16_t* __restrict__ Kb) {
  u32 idx = blockIdx.x * blockDim.x + threadIdx.x;  // B*H*S*64
  u32 i   = idx & 63;
  u32 s   = (idx >> 6) & (S_ - 1);
  u32 bh  = idx >> 17;  // b*H + h
  size_t base = ((size_t)bh * S_ + s) * HD_;
  float inv = __powf(10000.0f, -(float)(2 * i) / (float)HD_);
  float fr  = (float)s * inv;
  float cs, sn;
  __sincosf(fr, &cs, &sn);
  float x0 = (float)Qb[base + i], x1 = (float)Qb[base + i + 64];
  Qb[base + i]      = (bf16_t)(x0 * cs - x1 * sn);
  Qb[base + i + 64] = (bf16_t)(x1 * cs + x0 * sn);
  x0 = (float)Kb[base + i]; x1 = (float)Kb[base + i + 64];
  Kb[base + i]      = (bf16_t)(x0 * cs - x1 * sn);
  Kb[base + i + 64] = (bf16_t)(x1 * cs + x0 * sn);
}

// ---------------------------------------------------------------- attention
// One block = (b, h, 64 q-rows); 4 waves, each wave owns a 16-row Q strip.
// K tile staged via TDM; V tile staged manually (needs transpose).

#define KS_LD 136
#define VS_LD 40
#define PS_LD 40

__global__ __launch_bounds__(128) void k_attn(
    const bf16_t* __restrict__ Qb, const bf16_t* __restrict__ Kb,
    const bf16_t* __restrict__ Vb, bf16_t* __restrict__ AO) {
  __shared__ bf16_t Ks[32][KS_LD];    // [key][hd]
  __shared__ bf16_t VsT[HD_][VS_LD];  // [hd][key]
  __shared__ bf16_t Ps[4][16][PS_LD]; // per-wave P tile [row][key]

  const int t    = threadIdx.x;
  const int w    = t >> 5;
  const int lane = t & 31;
  const int ncol = lane & 15;
  const int half = lane >> 4;
  const int qt   = blockIdx.x;
  const int h    = blockIdx.y;
  const int b    = blockIdx.z;
  const size_t bh = ((size_t)b * H_ + h) * (size_t)S_ * HD_;
  const int q_base = qt * 64 + w * 16;
  const int wuid = __builtin_amdgcn_readfirstlane(t) >> 5;
  const u32 ldsK = (u32)(uintptr_t)&Ks[0][0];

  bf16x16 qf[4];
  {
    const bf16_t* qrow = Qb + bh + (size_t)(q_base + ncol) * HD_ + (half << 3);
#pragma unroll
    for (int f = 0; f < 4; ++f) {
      bf16x8 lo = *reinterpret_cast<const bf16x8*>(qrow + f * 32);
      bf16x8 hi = *reinterpret_cast<const bf16x8*>(qrow + f * 32 + 16);
#pragma unroll
      for (int i = 0; i < 8; ++i) { qf[f][i] = lo[i]; qf[f][8 + i] = hi[i]; }
    }
  }

  f32x8 acc[8];
#pragma unroll
  for (int f2 = 0; f2 < 8; ++f2) acc[f2] = zero8();
  float mi[8], li[8];
#pragma unroll
  for (int r = 0; r < 8; ++r) { mi[r] = -3.0e38f; li[r] = 0.0f; }

  const float scale = 0.08838834764831845f;  // 1/sqrt(128)
  const int nkb = qt * 2 + 2;                // causal: keys < (qt+1)*64

  for (int kb = 0; kb < nkb; ++kb) {
    const int k0 = kb * 32;
    if (wuid == 0) {  // TDM: K tile [32 keys][128 hd] -> Ks (pad to 136)
      tdm_load_2d(Kb + bh + (size_t)k0 * HD_, ldsK, HD_, S_ - k0, HD_, HD_, 32,
                  5, 3);
    }
    {  // manual V staging, transposed
      int vrow = t >> 4;
      int hc0  = (t & 15) * 8;
#pragma unroll
      for (int rr = 0; rr < 4; ++rr) {
        int kr = vrow + rr * 8;
        bf16x8 v = *reinterpret_cast<const bf16x8*>(
            Vb + bh + (size_t)(k0 + kr) * HD_ + hc0);
#pragma unroll
        for (int i = 0; i < 8; ++i) VsT[hc0 + i][kr] = v[i];
      }
    }
    if (wuid == 0) __builtin_amdgcn_s_wait_tensorcnt(0);
    __syncthreads();

    // S = Q * K^T  (two 16x16 tiles covering 32 keys)
    f32x8 sfr[2];
#pragma unroll
    for (int j = 0; j < 2; ++j) {
      f32x8 s = zero8();
#pragma unroll
      for (int f = 0; f < 4; ++f) {
        bf16x16 kf = frag_ld(&Ks[j * 16][f * 32], KS_LD);
        s = wmma_bf16(qf[f], kf, s);
      }
      sfr[j] = s;
    }

    float rmax[8];
#pragma unroll
    for (int r = 0; r < 8; ++r) {
      int qrow = q_base + r + 8 * half;
#pragma unroll
      for (int j = 0; j < 2; ++j) {
        int key = k0 + j * 16 + ncol;
        float sv = sfr[j][r] * scale;
        if (key > qrow) sv = -3.0e38f;
        sfr[j][r] = sv;
      }
      float mx = fmaxf(sfr[0][r], sfr[1][r]);
      mx = fmaxf(mx, __shfl_xor(mx, 1, 32));
      mx = fmaxf(mx, __shfl_xor(mx, 2, 32));
      mx = fmaxf(mx, __shfl_xor(mx, 4, 32));
      mx = fmaxf(mx, __shfl_xor(mx, 8, 32));
      rmax[r] = mx;
    }

#pragma unroll
    for (int r = 0; r < 8; ++r) {
      float mnew  = fmaxf(mi[r], rmax[r]);
      float alpha = __expf(mi[r] - mnew);
      mi[r] = mnew;
      float rs = 0.0f;
#pragma unroll
      for (int j = 0; j < 2; ++j) {
        float p = __expf(sfr[j][r] - mnew);
        sfr[j][r] = p;
        rs += p;
      }
      rs += __shfl_xor(rs, 1, 32);
      rs += __shfl_xor(rs, 2, 32);
      rs += __shfl_xor(rs, 4, 32);
      rs += __shfl_xor(rs, 8, 32);
      li[r] = li[r] * alpha + rs;
#pragma unroll
      for (int f2 = 0; f2 < 8; ++f2) acc[f2][r] *= alpha;
    }

    // C-layout -> A-layout transition for P via wave-private LDS
#pragma unroll
    for (int j = 0; j < 2; ++j)
#pragma unroll
      for (int r = 0; r < 8; ++r)
        Ps[w][r + 8 * half][j * 16 + ncol] = (bf16_t)sfr[j][r];

    bf16x16 pf = frag_ld(&Ps[w][0][0], PS_LD);
#pragma unroll
    for (int f2 = 0; f2 < 8; ++f2) {
      bf16x16 vf = frag_ld(&VsT[f2 * 16][0], VS_LD);
      acc[f2] = wmma_bf16(pf, vf, acc[f2]);
    }
    __syncthreads();
  }

#pragma unroll
  for (int f2 = 0; f2 < 8; ++f2)
#pragma unroll
    for (int r = 0; r < 8; ++r) {
      int qrow = q_base + r + 8 * half;
      float o = acc[f2][r] / li[r];
      AO[((size_t)b * S_ + qrow) * D_ + h * HD_ + f2 * 16 + ncol] = (bf16_t)o;
    }
}

// ---------------------------------------------------------------- out GEMM
// out[NTOK, D] fp32 = AO[NTOK, D] bf16 * Wo[D, D] (WoT pre-transposed)

__global__ __launch_bounds__(256) void k_gemm_out(
    const bf16_t* __restrict__ Abf, const bf16_t* __restrict__ WT,
    float* __restrict__ Cout) {
  __shared__ bf16_t As[2][128][LDA_];
  __shared__ bf16_t Bs[2][128][LDA_];

  const int t    = threadIdx.x;
  const int wid  = t >> 5;
  const int lane = t & 31;
  const int wm   = wid & 3;
  const int wn   = wid >> 2;
  const int m0   = blockIdx.y * 128;
  const int n0   = blockIdx.x * 128;
  const int wuid = __builtin_amdgcn_readfirstlane(t) >> 5;

  f32x8 acc[2][4];
#pragma unroll
  for (int mt = 0; mt < 2; ++mt)
#pragma unroll
    for (int nt = 0; nt < 4; ++nt) acc[mt][nt] = zero8();

  u32 ldsA[2] = { (u32)(uintptr_t)&As[0][0][0], (u32)(uintptr_t)&As[1][0][0] };
  u32 ldsB[2] = { (u32)(uintptr_t)&Bs[0][0][0], (u32)(uintptr_t)&Bs[1][0][0] };

  const int NIT = D_ / BK_;
  if (wuid == 0) {
    tdm_load_2d(Abf + (size_t)m0 * D_, ldsA[0], D_, NTOK - m0, D_, BK_, 128, 3, 3);
    tdm_load_2d(WT  + (size_t)n0 * D_, ldsB[0], D_, D_ - n0, D_, BK_, 128, 3, 3);
    __builtin_amdgcn_s_wait_tensorcnt(0);
  }
  __syncthreads();

  for (int it = 0; it < NIT; ++it) {
    const int cur = it & 1;
    if (wuid == 0 && it + 1 < NIT) {
      const int k1 = (it + 1) * BK_;
      tdm_load_2d(Abf + (size_t)m0 * D_ + k1, ldsA[cur ^ 1], D_, NTOK - m0, D_,
                  BK_, 128, 3, 3);
      tdm_load_2d(WT + (size_t)n0 * D_ + k1, ldsB[cur ^ 1], D_, D_ - n0, D_,
                  BK_, 128, 3, 3);
    }

    bf16x16 af[2], bfm[4];
#pragma unroll
    for (int mt = 0; mt < 2; ++mt)
      af[mt] = frag_ld(&As[cur][wm * 32 + mt * 16][0], LDA_);
#pragma unroll
    for (int nt = 0; nt < 4; ++nt)
      bfm[nt] = frag_ld(&Bs[cur][wn * 64 + nt * 16][0], LDA_);
#pragma unroll
    for (int mt = 0; mt < 2; ++mt)
#pragma unroll
      for (int nt = 0; nt < 4; ++nt)
        acc[mt][nt] = wmma_bf16(af[mt], bfm[nt], acc[mt][nt]);

    if (wuid == 0) __builtin_amdgcn_s_wait_tensorcnt(0);
    __syncthreads();
  }

  const u32 ncol = lane & 15;
  const u32 half = lane >> 4;
#pragma unroll
  for (int mt = 0; mt < 2; ++mt)
#pragma unroll
    for (int nt = 0; nt < 4; ++nt)
#pragma unroll
      for (int r = 0; r < 8; ++r) {
        u32 gm = (u32)m0 + wm * 32 + mt * 16 + r + 8 * half;
        u32 gn = (u32)n0 + wn * 64 + nt * 16 + ncol;
        Cout[(size_t)gm * D_ + gn] = acc[mt][nt][r];
      }
}

// ---------------------------------------------------------------- launch

extern "C" void kernel_launch(void* const* d_in, const int* in_sizes, int n_in,
                              void* d_out, int out_size, void* d_ws, size_t ws_size,
                              hipStream_t stream) {
  const float* hs   = (const float*)d_in[0];  // [B,S,D]
  const float* wqkv = (const float*)d_in[1];  // [D,3D]
  const float* wo   = (const float*)d_in[2];  // [D,D]
  float* out = (float*)d_out;                 // [B,S,D]

  char* ws = (char*)d_ws;
  size_t off = 0;
  bf16_t* Xbf  = (bf16_t*)(ws + off); off += (size_t)NTOK * D_ * sizeof(bf16_t);
  bf16_t* WqT  = (bf16_t*)(ws + off); off += (size_t)D_ * NQKV * sizeof(bf16_t);
  bf16_t* WoT  = (bf16_t*)(ws + off); off += (size_t)D_ * D_ * sizeof(bf16_t);
  bf16_t* Qb   = (bf16_t*)(ws + off); off += (size_t)NTOK * D_ * sizeof(bf16_t);
  bf16_t* Kb   = (bf16_t*)(ws + off); off += (size_t)NTOK * D_ * sizeof(bf16_t);
  bf16_t* Vb   = (bf16_t*)(ws + off); off += (size_t)NTOK * D_ * sizeof(bf16_t);
  bf16_t* AO   = (bf16_t*)(ws + off); off += (size_t)NTOK * D_ * sizeof(bf16_t);

  const int thr = 256;
  int nX = NTOK * D_;
  k_f32_to_bf16<<<(nX + thr - 1) / thr, thr, 0, stream>>>(hs, Xbf, nX);
  // weights converted + transposed so TDM tiles are row-contiguous
  k_f32_to_bf16_T<<<dim3(NQKV / thr, D_), thr, 0, stream>>>(wqkv, WqT, D_, NQKV);
  k_f32_to_bf16_T<<<dim3(D_ / thr, D_), thr, 0, stream>>>(wo, WoT, D_, D_);

  k_gemm_qkv<<<dim3(NQKV / 128, NTOK / 128), 256, 0, stream>>>(Xbf, WqT, Qb, Kb, Vb);

  int nR = B_ * H_ * S_ * 64;
  k_rope<<<nR / thr, thr, 0, stream>>>(Qb, Kb);

  k_attn<<<dim3(S_ / 64, H_, B_), 128, 0, stream>>>(Qb, Kb, Vb, AO);

  k_gemm_out<<<dim3(D_ / 128, NTOK / 128), 256, 0, stream>>>(AO, WoT, out);
}
